// GQA_77086073028878
// MI455X (gfx1250) — matrix operand order repaired and checked by reading, not compiled, not run on hardware
//
#include <hip/hip_runtime.h>
#include <stdint.h>

typedef __bf16 bf16;
typedef __attribute__((ext_vector_type(16))) __bf16 v16bf;
typedef __attribute__((ext_vector_type(8)))  float   v8f;

#define B_    2
#define T_    2048
#define DIM_  2048
#define H_    32
#define HKV_  8
#define HD_   64
#define NREP_ 4
#define MT_   (B_ * T_)          // 4096 rows
#define NEG_INF (-1.0e30f)

// ---- bf16 <-> f32 via native hardware converts (v_cvt_*, RTNE) ----
__device__ __forceinline__ bf16  f2bf(float f) { return (bf16)f; }
__device__ __forceinline__ float bf2f(bf16 b)  { return (float)b; }

// ---- 16-lane butterfly reductions: v_permlane16_b32 (VALU) with shfl fallback ----
#if __has_builtin(__builtin_amdgcn_permlane16)
__device__ __forceinline__ float pl16(float v, uint32_t s0, uint32_t s1) {
  int r = __builtin_amdgcn_permlane16(__float_as_int(v), __float_as_int(v),
                                      s0, s1, false, false);
  return __int_as_float(r);
}
__device__ __forceinline__ float redmax16(float v) {
  v = fmaxf(v, pl16(v, 0x67452301u, 0xEFCDAB89u));  // xor 1
  v = fmaxf(v, pl16(v, 0x54761032u, 0xDCFE98BAu));  // xor 2
  v = fmaxf(v, pl16(v, 0x32107654u, 0xBA98FEDCu));  // xor 4
  v = fmaxf(v, pl16(v, 0xFEDCBA98u, 0x76543210u));  // xor 8
  return v;
}
__device__ __forceinline__ float redsum16(float v) {
  v += pl16(v, 0x67452301u, 0xEFCDAB89u);
  v += pl16(v, 0x54761032u, 0xDCFE98BAu);
  v += pl16(v, 0x32107654u, 0xBA98FEDCu);
  v += pl16(v, 0xFEDCBA98u, 0x76543210u);
  return v;
}
#else
__device__ __forceinline__ float redmax16(float v) {
  for (int d = 1; d < 16; d <<= 1) v = fmaxf(v, __shfl_xor(v, d, 32));
  return v;
}
__device__ __forceinline__ float redsum16(float v) {
  for (int d = 1; d < 16; d <<= 1) v += __shfl_xor(v, d, 32);
  return v;
}
#endif

// ---- async global -> LDS (ASYNCcnt-tracked, CDNA5) ----
__device__ __forceinline__ void async_b128(uint32_t lds_off, const void* gaddr) {
  asm volatile("global_load_async_to_lds_b128 %0, %1, off"
               :: "v"(lds_off), "v"(gaddr) : "memory");
}
__device__ __forceinline__ void wait_async0() {
  asm volatile("s_wait_asynccnt 0x0" ::: "memory");
}
__device__ __forceinline__ uint32_t lds_off32(const void* p) {
  return (uint32_t)(uintptr_t)p;   // LDS aperture: addr[31:0] is the LDS byte offset
}

// ---- f32 -> bf16 cast ----
__global__ void cast_f32_bf16(const float* __restrict__ in, bf16* __restrict__ out, int n) {
  int i = blockIdx.x * blockDim.x + threadIdx.x;
  int stride = gridDim.x * blockDim.x;
  for (; i < n; i += stride) out[i] = f2bf(in[i]);
}

// ---- RoPE on interleaved pairs, in place on bf16 activations ----
__global__ void rope_kernel(bf16* __restrict__ qk, const float* __restrict__ fc,
                            const float* __restrict__ fs, int nheads) {
  int total = MT_ * nheads * (HD_ / 2);
  int i = blockIdx.x * blockDim.x + threadIdx.x;
  if (i >= total) return;
  int d    = i & 31;              // pair index within head (HD/2 = 32)
  int rest = i >> 5;
  int hh   = rest % nheads;
  int bt   = rest / nheads;       // b*T + t
  int t    = bt % T_;
  bf16* p = qk + (size_t)bt * nheads * HD_ + hh * HD_ + 2 * d;
  float xr = bf2f(p[0]), xi = bf2f(p[1]);
  float c = fc[t * 32 + d], s = fs[t * 32 + d];
  p[0] = f2bf(xr * c - xi * s);
  p[1] = f2bf(xr * s + xi * c);
}

// ---- Tiled bf16 WMMA GEMM: C[M,N] = A[M,K] * B[K,N], row-major ----
// Block tile 128x128x32. 8 waves (4x2), each wave 32x64 => 2x4 wmma tiles.
// A tile moved by async global->LDS b128 DMA; B tile register-transposed.
template <bool OUT_F32>
__global__ __launch_bounds__(256) void gemm_bf16_wmma(
    const bf16* __restrict__ A, const bf16* __restrict__ Bm,
    void* __restrict__ Cout, int M, int N, int K) {
  __shared__ __align__(16) bf16 As[128][40];   // [m][k], +8 pad
  __shared__ __align__(16) bf16 Bs[128][40];   // transposed: [n][k], +8 pad

  const int tid  = threadIdx.x;
  const int lane = tid & 31, wid = tid >> 5;
  const int h = lane >> 4, nl = lane & 15;     // wave halves / column lane
  const int wm0 = (wid >> 1) * 32, wn0 = (wid & 1) * 64;
  const int gm0 = blockIdx.y * 128, gn0 = blockIdx.x * 128;

  const int arow = tid >> 1, acs = (tid & 1) * 16;   // A tile loader (32B/thread)
  const int bk   = tid >> 3, bns = (tid & 7) * 16;   // B tile loader (32B/thread)

  const uint32_t lds_a = lds_off32(&As[arow][acs]);

  v8f c[2][4];
  #pragma unroll
  for (int i = 0; i < 2; ++i)
    #pragma unroll
    for (int j = 0; j < 4; ++j)
      #pragma unroll
      for (int r = 0; r < 8; ++r) c[i][j][r] = 0.0f;

  for (int k0 = 0; k0 < K; k0 += 32) {
    const bf16* agp = A + (size_t)(gm0 + arow) * K + k0 + acs;
    const uint4* bg = (const uint4*)(Bm + (size_t)(k0 + bk) * N + gn0 + bns);
    uint4 bv0 = bg[0], bv1 = bg[1];
    if (k0 + 32 < K)
      __builtin_prefetch(Bm + (size_t)(k0 + 32 + bk) * N + gn0 + bns, 0, 1);

    __syncthreads();   // previous iteration's fragment reads complete
    async_b128(lds_a,      agp);      // A tile: global -> LDS, no VGPR round-trip
    async_b128(lds_a + 16, agp + 8);
    {
      alignas(16) bf16 tmp[16];
      *(uint4*)&tmp[0] = bv0; *(uint4*)&tmp[8] = bv1;
      #pragma unroll
      for (int i = 0; i < 16; ++i) Bs[bns + i][bk] = tmp[i];
    }
    wait_async0();
    __syncthreads();   // tiles ready

    v16bf a[2], b[4];
    #pragma unroll
    for (int mt = 0; mt < 2; ++mt) {
      // A-frag (16x32 bf16): lane row M=nl; e<8 -> K=8h+e, e>=8 -> K=16+8h+(e-8)
      const bf16* ap = &As[wm0 + mt * 16 + nl][0];
      *((uint4*)&a[mt])     = *(const uint4*)(ap + h * 8);
      *((uint4*)&a[mt] + 1) = *(const uint4*)(ap + 16 + h * 8);
    }
    #pragma unroll
    for (int nt = 0; nt < 4; ++nt) {
      // B-frag (32x16 bf16): lane col N=nl; e<8 -> K=8h+e, e>=8 -> K=16+8h+(e-8)
      const bf16* bp = &Bs[wn0 + nt * 16 + nl][0];
      *((uint4*)&b[nt])     = *(const uint4*)(bp + h * 8);
      *((uint4*)&b[nt] + 1) = *(const uint4*)(bp + 16 + h * 8);
    }
    #pragma unroll
    for (int mt = 0; mt < 2; ++mt)
      #pragma unroll
      for (int nt = 0; nt < 4; ++nt)
        c[mt][nt] = __builtin_amdgcn_wmma_f32_16x16x32_bf16(
            false, a[mt], false, b[nt], (short)0, c[mt][nt], false, false);
  }

  #pragma unroll
  for (int mt = 0; mt < 2; ++mt)
    #pragma unroll
    for (int nt = 0; nt < 4; ++nt)
      #pragma unroll
      for (int r = 0; r < 8; ++r) {
        int row = gm0 + wm0 + mt * 16 + r + 8 * h;   // C layout: M = r + 8h
        int col = gn0 + wn0 + nt * 16 + nl;          // N = lane & 15
        float v = c[mt][nt][r];
        if (OUT_F32) ((float*)Cout)[(size_t)row * N + col] = v;
        else         ((bf16*)Cout)[(size_t)row * N + col] = f2bf(v);
      }
}

// ---- Flash attention (causal, GQA). Block = (b, q-head, 128 q rows); wave = 16 q rows.
__global__ __launch_bounds__(256) void attn_fwd(
    const bf16* __restrict__ Q, const bf16* __restrict__ Kc,
    const bf16* __restrict__ Vc, bf16* __restrict__ O) {
  __shared__ __align__(16) bf16 Ks[32][72];      // [key][hd], +8 pad
  __shared__ __align__(16) bf16 Vt[64][40];      // transposed: [hd][key], +8 pad
  __shared__ __align__(16) bf16 Ps[8][16][40];   // per-wave P staging: [row][key], +8 pad

  const int tid = threadIdx.x, lane = tid & 31, wid = tid >> 5;
  const int h = lane >> 4, nl = lane & 15;
  const int bh = blockIdx.x;
  const int bq = bh / H_, hq = bh % H_;
  const int hkv = hq / NREP_;
  const int qblock = blockIdx.y;
  const int q0w = qblock * 128 + wid * 16;

  // Persistent Q fragments (16 rows x HD=64 as 2 chunks of K=32)
  v16bf aq[2];
  {
    const bf16* qp = Q + (size_t)(bq * T_ + q0w + nl) * (H_ * HD_) + hq * HD_;
    #pragma unroll
    for (int ch = 0; ch < 2; ++ch) {
      *((uint4*)&aq[ch])     = *(const uint4*)(qp + ch * 32 + h * 8);
      *((uint4*)&aq[ch] + 1) = *(const uint4*)(qp + ch * 32 + 16 + h * 8);
    }
  }

  v8f acc[4];
  #pragma unroll
  for (int nt = 0; nt < 4; ++nt)
    #pragma unroll
    for (int r = 0; r < 8; ++r) acc[nt][r] = 0.0f;
  float m_r[8], l_r[8];
  #pragma unroll
  for (int r = 0; r < 8; ++r) { m_r[r] = NEG_INF; l_r[r] = 0.0f; }

  const int nblocks = qblock * 4 + 4;            // key blocks of 32, up to block diagonal
  const int kkey = tid >> 3, kseg = (tid & 7) * 8;
  const uint32_t lds_k = lds_off32(&Ks[kkey][kseg]);

  for (int j = 0; j < nblocks; ++j) {
    const int kb0 = j * 32;
    const bf16* kgp = Kc + (size_t)(bq * T_ + kb0 + kkey) * (HKV_ * HD_) + hkv * HD_ + kseg;
    uint4 vv = *(const uint4*)(Vc + (size_t)(bq * T_ + kb0 + kkey) * (HKV_ * HD_) + hkv * HD_ + kseg);
    __syncthreads();
    async_b128(lds_k, kgp);          // K tile: async global -> LDS (16B/thread)
    {
      alignas(16) bf16 tmp[8];
      *(uint4*)&tmp[0] = vv;
      #pragma unroll
      for (int i = 0; i < 8; ++i) Vt[kseg + i][kkey] = tmp[i];
    }
    wait_async0();
    __syncthreads();

    // S(16x32) = Q(16x64) * K^T : two 16-col tiles, K-dim 64 as 2 chunks
    v8f s0, s1;
    #pragma unroll
    for (int r = 0; r < 8; ++r) { s0[r] = 0.0f; s1[r] = 0.0f; }
    #pragma unroll
    for (int ch = 0; ch < 2; ++ch) {
      v16bf b0, b1;
      const bf16* kp0 = &Ks[nl][ch * 32 + h * 8];       // key tile 0: n = nl
      const bf16* kp1 = &Ks[16 + nl][ch * 32 + h * 8];  // key tile 1: n = 16+nl
      *((uint4*)&b0) = *(const uint4*)kp0; *((uint4*)&b0 + 1) = *(const uint4*)(kp0 + 16);
      *((uint4*)&b1) = *(const uint4*)kp1; *((uint4*)&b1 + 1) = *(const uint4*)(kp1 + 16);
      s0 = __builtin_amdgcn_wmma_f32_16x16x32_bf16(false, aq[ch], false, b0, (short)0, s0, false, false);
      s1 = __builtin_amdgcn_wmma_f32_16x16x32_bf16(false, aq[ch], false, b1, (short)0, s1, false, false);
    }

    // Online softmax per row (row M = r + 8h lives across 16 lanes of a half-wave)
    #pragma unroll
    for (int r = 0; r < 8; ++r) {
      int q = q0w + r + 8 * h;
      float v0 = s0[r] * 0.125f;                 // 1/sqrt(64)
      float v1 = s1[r] * 0.125f;
      if (kb0 + nl      > q) v0 = NEG_INF;
      if (kb0 + 16 + nl > q) v1 = NEG_INF;
      float mx = redmax16(fmaxf(v0, v1));
      float m_new = fmaxf(m_r[r], mx);
      float p0 = __expf(v0 - m_new);
      float p1 = __expf(v1 - m_new);
      float rs = redsum16(p0 + p1);
      float alpha = __expf(m_r[r] - m_new);
      l_r[r] = l_r[r] * alpha + rs;
      m_r[r] = m_new;
      #pragma unroll
      for (int nt = 0; nt < 4; ++nt) acc[nt][r] *= alpha;
      Ps[wid][r + 8 * h][nl]      = f2bf(p0);
      Ps[wid][r + 8 * h][16 + nl] = f2bf(p1);
    }
    __syncthreads();   // Ps (C-layout -> A-layout redistribution) visible

    // O(16x64) += P(16x32) * V(32x64)
    v16bf pa;
    const bf16* pp = &Ps[wid][nl][0];
    *((uint4*)&pa)     = *(const uint4*)(pp + h * 8);
    *((uint4*)&pa + 1) = *(const uint4*)(pp + 16 + h * 8);
    #pragma unroll
    for (int nt = 0; nt < 4; ++nt) {
      v16bf vb;
      const bf16* vp = &Vt[nt * 16 + nl][h * 8];   // e<8 -> key 8h+e, e>=8 -> key 16+8h+(e-8)
      *((uint4*)&vb)     = *(const uint4*)vp;
      *((uint4*)&vb + 1) = *(const uint4*)(vp + 16);
      acc[nt] = __builtin_amdgcn_wmma_f32_16x16x32_bf16(false, pa, false, vb, (short)0, acc[nt], false, false);
    }
  }

  #pragma unroll
  for (int nt = 0; nt < 4; ++nt)
    #pragma unroll
    for (int r = 0; r < 8; ++r) {
      int q = q0w + r + 8 * h;
      O[(size_t)(bq * T_ + q) * (H_ * HD_) + hq * HD_ + nt * 16 + nl] =
          f2bf(acc[nt][r] / l_r[r]);
    }
}

// ---- host-side orchestration ----
extern "C" void kernel_launch(void* const* d_in, const int* in_sizes, int n_in,
                              void* d_out, int out_size, void* d_ws, size_t ws_size,
                              hipStream_t stream) {
  const float* x  = (const float*)d_in[0];
  const float* fc = (const float*)d_in[1];
  const float* fs = (const float*)d_in[2];
  const float* wq = (const float*)d_in[3];
  const float* wk = (const float*)d_in[4];
  const float* wv = (const float*)d_in[5];
  const float* wo = (const float*)d_in[6];

  char* ws = (char*)d_ws;
  size_t off = 0;
  auto carve = [&](size_t elems) -> bf16* {
    bf16* p = (bf16*)(ws + off);
    off += ((elems * sizeof(bf16)) + 255) & ~(size_t)255;
    return p;
  };
  bf16* xb  = carve((size_t)MT_ * DIM_);
  bf16* wqb = carve((size_t)DIM_ * DIM_);
  bf16* wkb = carve((size_t)DIM_ * (HKV_ * HD_));
  bf16* wvb = carve((size_t)DIM_ * (HKV_ * HD_));
  bf16* wob = carve((size_t)DIM_ * DIM_);
  bf16* qb  = carve((size_t)MT_ * DIM_);
  bf16* kb  = carve((size_t)MT_ * (HKV_ * HD_));
  bf16* vb  = carve((size_t)MT_ * (HKV_ * HD_));
  bf16* ab  = carve((size_t)MT_ * DIM_);

  // 1) casts to bf16
  cast_f32_bf16<<<4096, 256, 0, stream>>>(x,  xb,  MT_ * DIM_);
  cast_f32_bf16<<<4096, 256, 0, stream>>>(wq, wqb, DIM_ * DIM_);
  cast_f32_bf16<<<2048, 256, 0, stream>>>(wk, wkb, DIM_ * (HKV_ * HD_));
  cast_f32_bf16<<<2048, 256, 0, stream>>>(wv, wvb, DIM_ * (HKV_ * HD_));
  cast_f32_bf16<<<4096, 256, 0, stream>>>(wo, wob, DIM_ * DIM_);

  // 2) QKV projections (WMMA bf16)
  dim3 gq(DIM_ / 128, MT_ / 128);                 // (16, 32)
  dim3 gkv((HKV_ * HD_) / 128, MT_ / 128);        // (4, 32)
  gemm_bf16_wmma<false><<<gq,  256, 0, stream>>>(xb, wqb, qb, MT_, DIM_,        DIM_);
  gemm_bf16_wmma<false><<<gkv, 256, 0, stream>>>(xb, wkb, kb, MT_, HKV_ * HD_,  DIM_);
  gemm_bf16_wmma<false><<<gkv, 256, 0, stream>>>(xb, wvb, vb, MT_, HKV_ * HD_,  DIM_);

  // 3) RoPE on Q and K
  rope_kernel<<<(MT_ * H_   * (HD_ / 2)) / 256, 256, 0, stream>>>(qb, fc, fs, H_);
  rope_kernel<<<(MT_ * HKV_ * (HD_ / 2)) / 256, 256, 0, stream>>>(kb, fc, fs, HKV_);

  // 4) Flash attention
  dim3 ga(B_ * H_, T_ / 128);                     // (64, 16)
  attn_fwd<<<ga, 256, 0, stream>>>(qb, kb, vb, ab);

  // 5) Output projection, f32 result to d_out
  gemm_bf16_wmma<true><<<gq, 256, 0, stream>>>(ab, wob, d_out, MT_, DIM_, DIM_);
}
